// HierMultiHeadAttn_42545946034840
// MI455X (gfx1250) — matrix operand order repaired and checked by reading, not compiled, not run on hardware
//
#include <hip/hip_runtime.h>

// ---------------------------------------------------------------------------
// Problem constants (from reference): B=4, T=2048, C=1024, H=16, DH=64,
// W0=1024, W1=256, W2=64, R=8, SCALING=2.0. stride=2.
// ---------------------------------------------------------------------------
#define B_ 4
#define T_ 2048
#define C_ 1024
#define H_ 16
#define DH_ 64

typedef __attribute__((ext_vector_type(16))) __bf16 v16bf;
typedef __attribute__((ext_vector_type(8)))  float  v8f;
typedef __attribute__((ext_vector_type(8)))  unsigned short ushort8;

union BFrag {
  v16bf   v;
  ushort8 h[2];
  unsigned short u[16];
};

__device__ inline unsigned short f2bf(float f) {
  union { float f; unsigned u; } cv; cv.f = f;
  unsigned u = cv.u;
  return (unsigned short)((u + 0x7FFFu + ((u >> 16) & 1u)) >> 16); // RNE
}

__device__ inline v8f zero8() {
  v8f z;
#pragma unroll
  for (int i = 0; i < 8; ++i) z[i] = 0.f;
  return z;
}

__device__ inline v8f wmma_bf16(const BFrag& a, const BFrag& b, v8f c) {
  return __builtin_amdgcn_wmma_f32_16x16x32_bf16(
      false, a.v, false, b.v, (short)0, c, false, false);
}

// A-fragment: 16x32 bf16, source row-major with leading dim ld (elements).
// base points at A[row0][k0]. Lane L: M = L&15; chunks at k = 8*(L>>4) and +16.
__device__ inline BFrag load_a(const unsigned short* base, size_t ld, int lane) {
  int lo = lane & 15, hi = lane >> 4;
  const unsigned short* p = base + (size_t)lo * ld + 8 * hi;
  BFrag f;
  f.h[0] = *(const ushort8*)(p);
  f.h[1] = *(const ushort8*)(p + 16);
  return f;
}

// B-fragment: 32x16 bf16 where B[k][n] = Src[n][k], Src row-major ld.
// base points at Src[n0][k0]. Lane L: n = L&15; chunks at k = 8*(L>>4) and +16.
__device__ inline BFrag load_b(const unsigned short* base, size_t ld, int lane) {
  int lo = lane & 15, hi = lane >> 4;
  const unsigned short* p = base + (size_t)lo * ld + 8 * hi;
  BFrag f;
  f.h[0] = *(const ushort8*)(p);
  f.h[1] = *(const ushort8*)(p + 16);
  return f;
}

// Async global->LDS copy of one 16-byte chunk (per lane), tracked by ASYNCcnt.
__device__ inline void async_copy_b128(unsigned lds_byte_addr,
                                       const void* global_addr) {
  asm volatile("global_load_async_to_lds_b128 %0, %1, off"
               :: "v"(lds_byte_addr), "v"(global_addr) : "memory");
}
__device__ inline void wait_async_le3() {
  asm volatile("s_wait_asynccnt 0x3" ::: "memory");
}
__device__ inline void wait_async_0() {
  asm volatile("s_wait_asynccnt 0x0" ::: "memory");
}

// ---------------------------------------------------------------------------
// Fold LoRA into weights: Deff = bf16(W + 2.0 * A @ B), W:(C,C) A:(C,8) B:(8,C)
// ---------------------------------------------------------------------------
__global__ __launch_bounds__(256) void fold_lora_kernel(
    const float* W0p, const float* A0p, const float* B0p,
    const float* W1p, const float* A1p, const float* B1p,
    const float* W2p, const float* A2p, const float* B2p,
    const float* W3p, const float* A3p, const float* B3p,
    unsigned short* D0, unsigned short* D1, unsigned short* D2, unsigned short* D3) {
  const float *W, *A, *Bm; unsigned short* D;
  switch (blockIdx.y) {
    case 0:  W = W0p; A = A0p; Bm = B0p; D = D0; break;
    case 1:  W = W1p; A = A1p; Bm = B1p; D = D1; break;
    case 2:  W = W2p; A = A2p; Bm = B2p; D = D2; break;
    default: W = W3p; A = A3p; Bm = B3p; D = D3; break;
  }
  size_t idx = (size_t)blockIdx.x * 256 + threadIdx.x; // over C_*C_
  int n = (int)(idx >> 10), k = (int)(idx & 1023);
  float acc = W[idx];
#pragma unroll
  for (int r = 0; r < 8; ++r)
    acc += 2.0f * A[(size_t)n * 8 + r] * Bm[(size_t)r * C_ + k];
  D[idx] = f2bf(acc);
}

// ---------------------------------------------------------------------------
// f32 -> bf16 cast, 4 elems/thread
// ---------------------------------------------------------------------------
__global__ __launch_bounds__(256) void cast_bf16_kernel(
    const float* src, unsigned short* dst, size_t n) {
  size_t i = ((size_t)blockIdx.x * 256 + threadIdx.x) * 4;
  if (i + 3 < n) {
#pragma unroll
    for (int j = 0; j < 4; ++j) dst[i + j] = f2bf(src[i + j]);
  }
}

// ---------------------------------------------------------------------------
// GEMM: Cf32/Cbf16[M,N] = Abf16[M,K] * Weff_bf16[N,K]^T   (WMMA bf16)
// Block 256 thr = 8 waves; WG tile 128Mx64N; wave tile 32x32 (2x2 of 16x16).
// Double-buffered LDS staging via GLOBAL_LOAD_ASYNC_TO_LDS_B128 (ASYNCcnt):
// per K-slab of 32, A-tile 128x32 (8KB) + B-tile 64x32 (4KB) are DMA'd to
// LDS (rows padded to 40 elems = 80B for bank spread), then fragments are
// ds_load'ed.  Each wave issues exactly 3 async ops per stage, so
// s_wait_asynccnt<=3 after issuing the next stage retires the current one
// (async loads complete in order).
// ---------------------------------------------------------------------------
__global__ __launch_bounds__(256) void gemm_bf16_kernel(
    const unsigned short* __restrict__ A, const unsigned short* __restrict__ W,
    float* Cf, unsigned short* Cb, int M, int N, int K) {
  constexpr int LDT = 40; // padded LDS row stride (elements)
  __shared__ unsigned short As[2][128 * LDT];
  __shared__ unsigned short Bs[2][64 * LDT];

  int tid = threadIdx.x, wave = tid >> 5, lane = tid & 31;
  int wm = wave >> 1, wn = wave & 1;
  int mBlock = blockIdx.y * 128, nBlock = blockIdx.x * 64;
  int mW = wm * 32, nW = wn * 32;

  v8f acc[2][2];
  acc[0][0] = zero8(); acc[0][1] = zero8();
  acc[1][0] = zero8(); acc[1][1] = zero8();

  auto stage = [&](int buf, int kc) {
    // A tile: 128 rows x 32 cols -> 512 chunks of 16B; 2 chunks per thread.
#pragma unroll
    for (int q = 0; q < 2; ++q) {
      int c = tid * 2 + q;
      int row = c >> 2, seg = c & 3;
      const unsigned short* g = A + (size_t)(mBlock + row) * K + kc + seg * 8;
      unsigned l = (unsigned)(size_t)&As[buf][row * LDT + seg * 8];
      async_copy_b128(l, g);
    }
    // B tile: 64 rows x 32 cols -> 256 chunks of 16B; 1 chunk per thread.
    {
      int row = tid >> 2, seg = tid & 3;
      const unsigned short* g = W + (size_t)(nBlock + row) * K + kc + seg * 8;
      unsigned l = (unsigned)(size_t)&Bs[buf][row * LDT + seg * 8];
      async_copy_b128(l, g);
    }
  };

  int nIter = K / 32;
  stage(0, 0);
  for (int i = 0; i < nIter; ++i) {
    int buf = i & 1;
    if (i + 1 < nIter) {
      stage(buf ^ 1, (i + 1) * 32);
      wait_async_le3();   // current buffer's 3 older async ops have retired
    } else {
      wait_async_0();
    }
    __syncthreads();      // LDS data visible to all waves

    const unsigned short* as = &As[buf][0];
    const unsigned short* bs = &Bs[buf][0];
    BFrag a0 = load_a(as + (size_t)(mW + 0)  * LDT, LDT, lane);
    BFrag a1 = load_a(as + (size_t)(mW + 16) * LDT, LDT, lane);
    BFrag b0 = load_b(bs + (size_t)(nW + 0)  * LDT, LDT, lane);
    BFrag b1 = load_b(bs + (size_t)(nW + 16) * LDT, LDT, lane);
    acc[0][0] = wmma_bf16(a0, b0, acc[0][0]);
    acc[0][1] = wmma_bf16(a0, b1, acc[0][1]);
    acc[1][0] = wmma_bf16(a1, b0, acc[1][0]);
    acc[1][1] = wmma_bf16(a1, b1, acc[1][1]);

    __syncthreads();      // everyone done reading buf before it is re-staged
  }

  int lo = lane & 15, hi = lane >> 4;
#pragma unroll
  for (int i = 0; i < 2; ++i)
#pragma unroll
    for (int j = 0; j < 2; ++j)
#pragma unroll
      for (int r = 0; r < 8; ++r) {
        int row = mBlock + mW + 16 * i + r + 8 * hi;
        int col = nBlock + nW + 16 * j + lo;
        float v = acc[i][j][r];
        size_t idx = (size_t)row * N + col;
        if (Cf) Cf[idx] = v;
        if (Cb) Cb[idx] = f2bf(v);
      }
}

// ---------------------------------------------------------------------------
// Pool: chunk pairs (stride 2) with learned query; optional f32/bf16 outputs
// of the kept suffix; accumulates sum((sv-tv)^2) into rsum.
// grid = (nChunks, B). Input rows (b, t, c) with batchTok tokens per batch.
// ---------------------------------------------------------------------------
__global__ __launch_bounds__(256) void pool_kernel(
    const float* Kin, const float* Vin, const float* q,
    float* Kf_out, float* Vf_out, unsigned short* Kb_out, unsigned short* Vb_out,
    float* rsum, int batchTok, int keepFrom, int keepCount) {
  __shared__ float red[256];
  int tid = threadIdx.x;
  int n = blockIdx.x, b = blockIdx.y;
  const float* k0 = Kin + ((size_t)b * batchTok + 2 * n) * C_;
  const float* k1 = k0 + C_;
  const float* v0 = Vin + ((size_t)b * batchTok + 2 * n) * C_;
  const float* v1 = v0 + C_;

  float s0 = 0.f, s1 = 0.f;
  for (int c = tid; c < C_; c += 256) {
    float qc = q[c];
    s0 += qc * k0[c];
    s1 += qc * k1[c];
  }
  red[tid] = s0; __syncthreads();
  for (int s = 128; s > 0; s >>= 1) { if (tid < s) red[tid] += red[tid + s]; __syncthreads(); }
  s0 = red[0]; __syncthreads();
  red[tid] = s1; __syncthreads();
  for (int s = 128; s > 0; s >>= 1) { if (tid < s) red[tid] += red[tid + s]; __syncthreads(); }
  s1 = red[0]; __syncthreads();

  float mx = fmaxf(s0, s1);
  float e0 = __expf(s0 - mx), e1 = __expf(s1 - mx);
  float inv = 1.f / (e0 + e1);
  float w0 = e0 * inv, w1 = e1 * inv;

  float rp = 0.f;
  bool keep = (n >= keepFrom);
  size_t outBase = keep ? ((size_t)b * keepCount + (n - keepFrom)) * C_ : 0;
  for (int c = tid; c < C_; c += 256) {
    float ka = k0[c], kb = k1[c], va = v0[c], vb = v1[c];
    float sk = w0 * ka + w1 * kb;
    float sv = w0 * va + w1 * vb;
    float tv = 0.5f * (va + vb);
    float d = sv - tv;
    rp += d * d;
    if (keep) {
      if (Kf_out) Kf_out[outBase + c] = sk;
      if (Vf_out) Vf_out[outBase + c] = sv;
      if (Kb_out) Kb_out[outBase + c] = f2bf(sk);
      if (Vb_out) Vb_out[outBase + c] = f2bf(sv);
    }
  }
  red[tid] = rp; __syncthreads();
  for (int s = 128; s > 0; s >>= 1) { if (tid < s) red[tid] += red[tid + s]; __syncthreads(); }
  if (tid == 0) atomicAdd(rsum, red[0]);
}

// ---------------------------------------------------------------------------
// bf16 transpose (b,t,c) -> (b,c,t)  for V so attention B-frags are contiguous
// ---------------------------------------------------------------------------
__global__ __launch_bounds__(256) void transpose_v_kernel(
    const unsigned short* src, unsigned short* dst,
    int Tm, int srcTokOff, int srcBatchStride) {
  size_t idx = (size_t)blockIdx.x * 256 + threadIdx.x; // over Tm*C_
  int b = blockIdx.y;
  int t = (int)(idx / C_), c = (int)(idx % C_);
  dst[((size_t)b * C_ + c) * Tm + t] =
      src[(size_t)b * srcBatchStride + (size_t)(srcTokOff + t) * C_ + c];
}

// ---------------------------------------------------------------------------
// Flash attention (non-causal), one level, WMMA bf16, f32 softmax.
// Block 128 thr = 4 waves; wave handles 16 queries of one (b,h).
// S^T = K x Q^T so softmax stats are per-lane (query = lane&15); the exp'd
// S^T C-fragments map locally (no shuffles) onto the P A-fragment for P x V.
// Writes/accumulates lw[level] * O into Oacc (b,t,c) f32.
// ---------------------------------------------------------------------------
__global__ __launch_bounds__(128) void attn_kernel(
    const unsigned short* Qb, const unsigned short* Kb, const unsigned short* Vt,
    float* Oacc, const float* logits, int level, int Tm,
    long kBatchStride, int kTokOff, int accumulate) {
  int b = blockIdx.z, h = blockIdx.y;
  int wave = threadIdx.x >> 5, lane = threadIdx.x & 31;
  int lo = lane & 15, hi = lane >> 4;
  int t0 = (blockIdx.x * 4 + wave) * 16;

  float g0 = logits[0], g1 = logits[1], g2 = logits[2];
  float gm = fmaxf(g0, fmaxf(g1, g2));
  float x0 = __expf(g0 - gm), x1 = __expf(g1 - gm), x2 = __expf(g2 - gm);
  float lw = (level == 0 ? x0 : level == 1 ? x1 : x2) / (x0 + x1 + x2);
  const float scale = 0.125f; // DH^-0.5

  const unsigned short* qbase = Qb + ((size_t)b * T_ + t0) * C_ + h * DH_;
  BFrag qf0 = load_b(qbase,      C_, lane);   // d 0..31
  BFrag qf1 = load_b(qbase + 32, C_, lane);   // d 32..63

  float m = -3.0e38f, l = 0.f;
  v8f o[4];
#pragma unroll
  for (int j = 0; j < 4; ++j) o[j] = zero8();

  for (int kc = 0; kc < Tm; kc += 32) {
    const unsigned short* kbase =
        Kb + (size_t)b * kBatchStride + (size_t)(kTokOff + kc) * C_ + h * DH_;
    if (kc + 32 < Tm)
      __builtin_prefetch(kbase + 32 * C_, 0, 1);
    BFrag k00 = load_a(kbase,                 C_, lane); // keys kc..+15, d0..31
    BFrag k01 = load_a(kbase + 32,            C_, lane);
    BFrag k10 = load_a(kbase + 16 * C_,       C_, lane); // keys kc+16..+31
    BFrag k11 = load_a(kbase + 16 * C_ + 32,  C_, lane);

    v8f s0 = zero8(), s1 = zero8();
    s0 = wmma_bf16(k00, qf0, s0);
    s0 = wmma_bf16(k01, qf1, s0);
    s1 = wmma_bf16(k10, qf0, s1);
    s1 = wmma_bf16(k11, qf1, s1);

    float p0[8], p1[8];
    float mx = -3.0e38f;
#pragma unroll
    for (int r = 0; r < 8; ++r) {
      p0[r] = s0[r] * scale; mx = fmaxf(mx, p0[r]);
      p1[r] = s1[r] * scale; mx = fmaxf(mx, p1[r]);
    }
    mx = fmaxf(mx, __shfl_xor(mx, 16, 32)); // combine lane-pair: full 32 keys
    float mnew = fmaxf(m, mx);
    float alpha = __expf(m - mnew);

    BFrag pf;
    float ls = 0.f;
#pragma unroll
    for (int r = 0; r < 8; ++r) {
      float e0v = __expf(p0[r] - mnew);
      float e1v = __expf(p1[r] - mnew);
      ls += e0v + e1v;
      pf.u[r]     = f2bf(e0v);  // P A-frag: local mapping, no shuffles
      pf.u[8 + r] = f2bf(e1v);
    }
    ls += __shfl_xor(ls, 16, 32);
    l = l * alpha + ls;
    m = mnew;

    float af[8];
#pragma unroll
    for (int r = 0; r < 8; ++r) af[r] = __shfl(alpha, r + 8 * hi, 32);

    const unsigned short* vbase = Vt + ((size_t)b * C_ + h * DH_) * Tm + kc;
#pragma unroll
    for (int j = 0; j < 4; ++j) {
#pragma unroll
      for (int r = 0; r < 8; ++r) o[j][r] *= af[r];
      BFrag vf = load_b(vbase + (size_t)(16 * j) * Tm, Tm, lane);
      o[j] = wmma_bf16(pf, vf, o[j]);
    }
  }

  float lf[8];
#pragma unroll
  for (int r = 0; r < 8; ++r) {
    float li = __shfl(l, r + 8 * hi, 32);
    lf[r] = (li > 0.f) ? (1.f / li) : 0.f;
  }
#pragma unroll
  for (int j = 0; j < 4; ++j)
#pragma unroll
    for (int r = 0; r < 8; ++r) {
      float val = o[j][r] * lf[r] * lw;
      size_t idx = ((size_t)b * T_ + t0 + r + 8 * hi) * C_ + h * DH_ + 16 * j + lo;
      if (accumulate) Oacc[idx] += val; else Oacc[idx] = val;
    }
}

__global__ void init_rsum_kernel(float* rsum) {
  if (threadIdx.x < 2) rsum[threadIdx.x] = 0.f;
}

__global__ void finalize_recon_kernel(const float* rsum, float* out) {
  // recon = r1 + 0.5*128*r2 ; r1 = sum1/(4*512*1024), r2 = sum2/(4*128*1024)
  out[0] = rsum[0] * (1.f / (4.f * 512.f * 1024.f)) +
           64.f * rsum[1] * (1.f / (4.f * 128.f * 1024.f));
}

// ---------------------------------------------------------------------------
extern "C" void kernel_launch(void* const* d_in, const int* in_sizes, int n_in,
                              void* d_out, int out_size, void* d_ws, size_t ws_size,
                              hipStream_t stream) {
  (void)in_sizes; (void)n_in; (void)out_size; (void)ws_size;
  const float* x       = (const float*)d_in[0];
  const float* Wq_w    = (const float*)d_in[1];
  const float* Wq_A    = (const float*)d_in[2];
  const float* Wq_B    = (const float*)d_in[3];
  const float* Wk_w    = (const float*)d_in[4];
  const float* Wk_A    = (const float*)d_in[5];
  const float* Wk_B    = (const float*)d_in[6];
  const float* Wv_w    = (const float*)d_in[7];
  const float* Wv_A    = (const float*)d_in[8];
  const float* Wv_B    = (const float*)d_in[9];
  const float* Wo_w    = (const float*)d_in[10];
  const float* Wo_A    = (const float*)d_in[11];
  const float* Wo_B    = (const float*)d_in[12];
  const float* q_pool  = (const float*)d_in[13];
  const float* logits  = (const float*)d_in[14];

  float* out = (float*)d_out; // [B*T*C] + [1] recon

  const size_t MT = (size_t)B_ * T_;          // 8192 rows
  char* w = (char*)d_ws;
  size_t off = 0;
  auto alloc = [&](size_t bytes) -> void* {
    void* p = w + off; off += (bytes + 255) & ~(size_t)255; return p;
  };
  unsigned short* xb   = (unsigned short*)alloc(MT * C_ * 2);      // x bf16
  unsigned short* Wq_e = (unsigned short*)alloc((size_t)C_ * C_ * 2);
  unsigned short* Wk_e = (unsigned short*)alloc((size_t)C_ * C_ * 2);
  unsigned short* Wv_e = (unsigned short*)alloc((size_t)C_ * C_ * 2);
  unsigned short* Wo_e = (unsigned short*)alloc((size_t)C_ * C_ * 2);
  unsigned short* Qb   = (unsigned short*)alloc(MT * C_ * 2);
  unsigned short* Kb   = (unsigned short*)alloc(MT * C_ * 2);
  unsigned short* Vb   = (unsigned short*)alloc(MT * C_ * 2);
  float*          Kf   = (float*)alloc(MT * C_ * 4);
  float*          Vf   = (float*)alloc(MT * C_ * 4);
  float*          K1f  = (float*)alloc((size_t)B_ * 256 * C_ * 4);
  float*          V1f  = (float*)alloc((size_t)B_ * 256 * C_ * 4);
  unsigned short* K1b  = (unsigned short*)alloc((size_t)B_ * 256 * C_ * 2);
  unsigned short* V1b  = (unsigned short*)alloc((size_t)B_ * 256 * C_ * 2);
  unsigned short* K2b  = (unsigned short*)alloc((size_t)B_ * 64 * C_ * 2);
  unsigned short* V2b  = (unsigned short*)alloc((size_t)B_ * 64 * C_ * 2);
  unsigned short* Vt0  = (unsigned short*)alloc((size_t)B_ * C_ * 1024 * 2);
  unsigned short* Vt1  = (unsigned short*)alloc((size_t)B_ * C_ * 256 * 2);
  unsigned short* Vt2  = (unsigned short*)alloc((size_t)B_ * C_ * 64 * 2);
  float*          Oacc = (float*)alloc(MT * C_ * 4);
  unsigned short* attnb= (unsigned short*)alloc(MT * C_ * 2);
  float*          rsum = (float*)alloc(256);

  init_rsum_kernel<<<1, 32, 0, stream>>>(rsum);

  // Fold LoRA into effective bf16 weights.
  fold_lora_kernel<<<dim3((C_ * C_) / 256, 4), 256, 0, stream>>>(
      Wq_w, Wq_A, Wq_B, Wk_w, Wk_A, Wk_B, Wv_w, Wv_A, Wv_B, Wo_w, Wo_A, Wo_B,
      Wq_e, Wk_e, Wv_e, Wo_e);

  // x -> bf16
  cast_bf16_kernel<<<(unsigned)(MT * C_ / 1024), 256, 0, stream>>>(x, xb, MT * C_);

  // Q/K/V projections (Q needs only bf16; K/V dual-store f32 + bf16).
  dim3 ggrid(C_ / 64, (unsigned)(MT / 128));
  gemm_bf16_kernel<<<ggrid, 256, 0, stream>>>(xb, Wq_e, nullptr, Qb, (int)MT, C_, C_);
  gemm_bf16_kernel<<<ggrid, 256, 0, stream>>>(xb, Wk_e, Kf, Kb, (int)MT, C_, C_);
  gemm_bf16_kernel<<<ggrid, 256, 0, stream>>>(xb, Wv_e, Vf, Vb, (int)MT, C_, C_);

  // Level-1 pooling over first 1024 tokens (512 chunks of 2); keep last 256.
  pool_kernel<<<dim3(512, B_), 256, 0, stream>>>(
      Kf, Vf, q_pool, K1f, V1f, K1b, V1b, rsum + 0, T_, 256, 256);
  // Level-2 pooling over K1/V1 (128 pairs); keep last 64 (bf16 only).
  pool_kernel<<<dim3(128, B_), 256, 0, stream>>>(
      K1f, V1f, q_pool + C_, nullptr, nullptr, K2b, V2b, rsum + 1, 256, 64, 64);

  // V transposes (b,t,c) -> (b,c,t) for attention B-fragments.
  transpose_v_kernel<<<dim3((1024u * C_) / 256, B_), 256, 0, stream>>>(
      Vb, Vt0, 1024, 1024, T_ * C_);
  transpose_v_kernel<<<dim3((256u * C_) / 256, B_), 256, 0, stream>>>(
      V1b, Vt1, 256, 0, 256 * C_);
  transpose_v_kernel<<<dim3((64u * C_) / 256, B_), 256, 0, stream>>>(
      V2b, Vt2, 64, 0, 64 * C_);

  // Three attention levels, accumulating lw[level]*o into Oacc.
  dim3 agrid(T_ / 64, H_, B_); // 4 waves/block * 16 queries
  attn_kernel<<<agrid, 128, 0, stream>>>(
      Qb, Kb, Vt0, Oacc, logits, 0, 1024, (long)T_ * C_, 1024, 0);
  attn_kernel<<<agrid, 128, 0, stream>>>(
      Qb, K1b, Vt1, Oacc, logits, 1, 256, (long)256 * C_, 0, 1);
  attn_kernel<<<agrid, 128, 0, stream>>>(
      Qb, K2b, Vt2, Oacc, logits, 2, 64, (long)64 * C_, 0, 1);

  // Output projection: out = Oacc @ Wo_eff^T  (f32 result straight to d_out).
  cast_bf16_kernel<<<(unsigned)(MT * C_ / 1024), 256, 0, stream>>>(Oacc, attnb, MT * C_);
  gemm_bf16_kernel<<<ggrid, 256, 0, stream>>>(attnb, Wo_e, out, nullptr, (int)MT, C_, C_);

  finalize_recon_kernel<<<1, 1, 0, stream>>>(rsum, out + MT * C_);
}
